// EquiAttention_42210938585550
// MI455X (gfx1250) — matrix operand order repaired and checked by reading, not compiled, not run on hardware
//
#include <hip/hip_runtime.h>

// ---------------------------------------------------------------------------
// EquiAttention for MI455X (gfx1250, wave32, WMMA)
//   B=64, N=1024, IN_V=16, OUT_V=16, HS=64, INCR=2  ->  d_qk=192, d_v=64
// Kernel 1: per-token projections -> f16 Q[B,N,192], K[B,N,192], Vt[B,64,N]
// Kernel 2: flash attention; K/V tiles staged to LDS via async DMA
//           (global_load_async_to_lds_b128 + s_wait_asynccnt), double-buffered,
//           WMMA fed from LDS with ds_load_b128.
// ---------------------------------------------------------------------------

typedef __attribute__((ext_vector_type(16))) _Float16 v16h;
typedef __attribute__((ext_vector_type(8)))  float    v8f;

union F16Frag { v16h h; unsigned u[8]; };

#define BATCH 64
#define NTOK  1024
#define DQK   192
#define DV    64

// ---------------------------------------------------------------------------
// Prepass: normalize 4-vectors, project, emit f16 Q/K/Vt
// ---------------------------------------------------------------------------
__global__ __launch_bounds__(256)
void equi_prepass(const float* __restrict__ vectors, const float* __restrict__ scalars,
                  const float* __restrict__ Wq,  const float* __restrict__ Wq_s,
                  const float* __restrict__ bq_s,
                  const float* __restrict__ Wk,  const float* __restrict__ Wk_s,
                  const float* __restrict__ bk_s,
                  const float* __restrict__ Wv,
                  _Float16* __restrict__ qb, _Float16* __restrict__ kb,
                  _Float16* __restrict__ vtb)
{
    __shared__ float sWq[32 * 16];
    __shared__ float sWk[32 * 16];
    __shared__ float sWv[16 * 16];   // exp() applied at load
    __shared__ float sWqs[64 * 64];
    __shared__ float sWks[64 * 64];
    __shared__ float sb[128];

    const int tid = threadIdx.x;
    for (int i = tid; i < 512; i += 256) { sWq[i] = Wq[i]; sWk[i] = Wk[i]; }
    if (tid < 256) sWv[tid] = __expf(Wv[tid]);
    for (int i = tid; i < 4096; i += 256) { sWqs[i] = Wq_s[i]; sWks[i] = Wk_s[i]; }
    if (tid < 64) { sb[tid] = bq_s[tid]; sb[64 + tid] = bk_s[tid]; }
    __syncthreads();

    const int t = blockIdx.x * 256 + tid;        // token index
    const int b = t >> 10;
    const int n = t & 1023;

    // ---- load & Lorentz-normalize the 16 four-vectors ----
    float vec[64];
    const float* vp = vectors + (size_t)t * 64;
    #pragma unroll
    for (int j = 0; j < 16; ++j) {
        float a0 = vp[j * 4 + 0], a1 = vp[j * 4 + 1];
        float a2 = vp[j * 4 + 2], a3 = vp[j * 4 + 3];
        float nrm = a0 * a0 - a1 * a1 - a2 * a2 - a3 * a3;
        float inv = rsqrtf(fmaxf(fabsf(nrm), 1e-5f));
        vec[j * 4 + 0] = a0 * inv; vec[j * 4 + 1] = a1 * inv;
        vec[j * 4 + 2] = a2 * inv; vec[j * 4 + 3] = a3 * inv;
    }

    const float scale = 0.07216878364870322f;    // 1/sqrt(192), folded into Q
    _Float16* qo = qb + (size_t)t * DQK;
    _Float16* ko = kb + (size_t)t * DQK;

    // ---- Q/K vector projections (32x16), metric folded into K ----
    #pragma unroll 4
    for (int i = 0; i < 32; ++i) {
        float q0 = 0.f, q1 = 0.f, q2 = 0.f, q3 = 0.f;
        float k0 = 0.f, k1 = 0.f, k2 = 0.f, k3 = 0.f;
        #pragma unroll
        for (int j = 0; j < 16; ++j) {
            float wq = sWq[i * 16 + j], wk = sWk[i * 16 + j];
            q0 += wq * vec[j * 4 + 0]; q1 += wq * vec[j * 4 + 1];
            q2 += wq * vec[j * 4 + 2]; q3 += wq * vec[j * 4 + 3];
            k0 += wk * vec[j * 4 + 0]; k1 += wk * vec[j * 4 + 1];
            k2 += wk * vec[j * 4 + 2]; k3 += wk * vec[j * 4 + 3];
        }
        qo[i * 4 + 0] = (_Float16)(q0 * scale);
        qo[i * 4 + 1] = (_Float16)(q1 * scale);
        qo[i * 4 + 2] = (_Float16)(q2 * scale);
        qo[i * 4 + 3] = (_Float16)(q3 * scale);
        ko[i * 4 + 0] = (_Float16)( k0);         // metric (+,-,-,-)
        ko[i * 4 + 1] = (_Float16)(-k1);
        ko[i * 4 + 2] = (_Float16)(-k2);
        ko[i * 4 + 3] = (_Float16)(-k3);
    }

    // ---- V projection with exp(Wv), stored transposed: Vt[b][dim][n] ----
    _Float16* vo = vtb + (size_t)b * DV * NTOK + n;
    #pragma unroll 4
    for (int i = 0; i < 16; ++i) {
        float a0 = 0.f, a1 = 0.f, a2 = 0.f, a3 = 0.f;
        #pragma unroll
        for (int j = 0; j < 16; ++j) {
            float w = sWv[i * 16 + j];
            a0 += w * vec[j * 4 + 0]; a1 += w * vec[j * 4 + 1];
            a2 += w * vec[j * 4 + 2]; a3 += w * vec[j * 4 + 3];
        }
        vo[(size_t)(i * 4 + 0) * NTOK] = (_Float16)a0;
        vo[(size_t)(i * 4 + 1) * NTOK] = (_Float16)a1;
        vo[(size_t)(i * 4 + 2) * NTOK] = (_Float16)a2;
        vo[(size_t)(i * 4 + 3) * NTOK] = (_Float16)a3;
    }

    // ---- scalar projections (64x64 + bias) ----
    float s[64];
    const float* sp = scalars + (size_t)t * 64;
    #pragma unroll
    for (int j = 0; j < 64; ++j) s[j] = sp[j];
    #pragma unroll 2
    for (int h = 0; h < 64; ++h) {
        float aq = sb[h], ak = sb[64 + h];
        #pragma unroll
        for (int j = 0; j < 64; ++j) {
            aq += sWqs[h * 64 + j] * s[j];
            ak += sWks[h * 64 + j] * s[j];
        }
        qo[128 + h] = (_Float16)(aq * scale);
        ko[128 + h] = (_Float16)ak;
    }
}

// ---------------------------------------------------------------------------
// Flash attention: 8 waves/block, 16 query rows per wave, 32 keys per step.
// K/V tiles DMA'd to LDS (async, double-buffered); 16 WMMA per step.
// ---------------------------------------------------------------------------
#define KTILE_BYTES (32 * DQK * 2)     // 12288
#define VTILE_BYTES (DV * 32 * 2)      // 4096

__global__ __launch_bounds__(256)
void equi_attn(const _Float16* __restrict__ qb, const _Float16* __restrict__ kb,
               const _Float16* __restrict__ vtb, float* __restrict__ out)
{
    __shared__ _Float16 ktile[2][32 * DQK];      // 2 x 12 KB, row-major [key][d]
    __shared__ _Float16 vtile[2][DV * 32];       // 2 x 4 KB,  row-major [dim][key]
    __shared__ _Float16 pshm[8][16 * 32];        // per-wave P staging (C->A layout)

    const int tid  = threadIdx.x;
    const int lane = tid & 31;
    const int wave = tid >> 5;
    const int hf   = lane >> 4;                  // half-wave id
    const int lm   = lane & 15;
    const int b    = blockIdx.x >> 3;
    const int q0r  = (blockIdx.x & 7) * 128 + wave * 16;

    const _Float16* kbase = kb  + (size_t)b * NTOK * DQK;
    const _Float16* vbase = vtb + (size_t)b * DV * NTOK;

    // LDS byte addresses (low 32 bits of the flat shared pointer = LDS offset)
    const unsigned kt_lds = (unsigned)(uintptr_t)&ktile[0][0];
    const unsigned vt_lds = (unsigned)(uintptr_t)&vtile[0][0];

    // --- async stage of one 32-key K tile + V tile into LDS buffer bufi ---
    auto stage = [&](int bufi, int kv) {
        const _Float16* gk = kbase + (size_t)kv * DQK;   // 24576 B contiguous
        #pragma unroll
        for (int i = 0; i < 6; ++i) {
            int e = (tid + i * 256) * 8;                 // half index (16B chunks)
            const _Float16* g = gk + e;
            unsigned l = kt_lds + (unsigned)(bufi * KTILE_BYTES + e * 2);
            asm volatile("global_load_async_to_lds_b128 %0, %1, off"
                         :: "v"(l), "v"(g) : "memory");
        }
        {   // V tile: 64 rows x 32 keys; 64 B per row from Vt[b][row][kv..]
            int row = tid >> 2, seg = tid & 3;
            const _Float16* g = vbase + (size_t)row * NTOK + kv + seg * 8;
            unsigned l = vt_lds + (unsigned)(bufi * VTILE_BYTES + (row * 32 + seg * 8) * 2);
            asm volatile("global_load_async_to_lds_b128 %0, %1, off"
                         :: "v"(l), "v"(g) : "memory");
        }
    };

    // ---- Q tile: 16 rows x 192, six 16x32 A-fragments (global, read once) ----
    // A operand, contiguous k order used here: M = lane%16 ; K = half*16 + v*2
    F16Frag qf[6];
    const _Float16* qrow = qb + ((size_t)b * NTOK + q0r + lm) * DQK;
    #pragma unroll
    for (int f = 0; f < 6; ++f) {
        #pragma unroll
        for (int v = 0; v < 8; ++v) {
            int d = f * 32 + hf * 16 + v * 2;
            qf[f].u[v] = *(const unsigned*)(qrow + d);
        }
    }

    v8f o0 = {}, o1 = {}, o2 = {}, o3 = {};
    float m[8], l[8];
    #pragma unroll
    for (int g = 0; g < 8; ++g) { m[g] = -1e30f; l[g] = 0.f; }

    _Float16* ps = &pshm[wave][0];

    stage(0, 0);                                  // prologue DMA

    for (int it = 0; it < NTOK / 32; ++it) {
        const int  kv   = it * 32;
        const int  bufi = it & 1;

        asm volatile("s_wait_asynccnt 0" ::: "memory");
        __syncthreads();                          // tile ready; prev buffer free
        if (it + 1 < NTOK / 32) stage(bufi ^ 1, kv + 32);

        const _Float16* kbuf = &ktile[bufi][0];
        const _Float16* vbuf = &vtile[bufi][0];

        // ---- scores: two 16x16 tiles over d=192 (B-fragments from LDS) ----
        v8f s0 = {}, s1 = {};
        #pragma unroll
        for (int f = 0; f < 6; ++f) {
            // B operand, same contiguous k order as A: N = lane%16 ; K = half*16 + v*2
            F16Frag kb0, kb1;
            const unsigned* k0 = (const unsigned*)(kbuf + (size_t)lm * DQK + f * 32 + hf * 16);
            const unsigned* k1 = (const unsigned*)(kbuf + (size_t)(16 + lm) * DQK + f * 32 + hf * 16);
            #pragma unroll
            for (int v = 0; v < 8; ++v) { kb0.u[v] = k0[v]; kb1.u[v] = k1[v]; }
            s0 = __builtin_amdgcn_wmma_f32_16x16x32_f16(false, qf[f].h, false, kb0.h,
                                                        (short)0, s0, false, false);
            s1 = __builtin_amdgcn_wmma_f32_16x16x32_f16(false, qf[f].h, false, kb1.h,
                                                        (short)0, s1, false, false);
        }

        // ---- online softmax (rows at C-frag [g], cols across 16 lanes) ----
        float p0[8], p1[8];
        #pragma unroll
        for (int g = 0; g < 8; ++g) {
            float x = fmaxf(s0[g], s1[g]);
            x = fmaxf(x, __shfl_xor(x, 1, 16));
            x = fmaxf(x, __shfl_xor(x, 2, 16));
            x = fmaxf(x, __shfl_xor(x, 4, 16));
            x = fmaxf(x, __shfl_xor(x, 8, 16));
            float mn = fmaxf(m[g], x);
            float alpha = __expf(m[g] - mn);
            m[g] = mn;
            p0[g] = __expf(s0[g] - mn);
            p1[g] = __expf(s1[g] - mn);
            float rs = p0[g] + p1[g];
            rs += __shfl_xor(rs, 1, 16);
            rs += __shfl_xor(rs, 2, 16);
            rs += __shfl_xor(rs, 4, 16);
            rs += __shfl_xor(rs, 8, 16);
            l[g] = l[g] * alpha + rs;
            o0[g] *= alpha; o1[g] *= alpha; o2[g] *= alpha; o3[g] *= alpha;
        }

        // ---- P: C-layout f32 -> LDS -> A-layout f16 (intra-wave, ds in-order) ----
        #pragma unroll
        for (int g = 0; g < 8; ++g) {
            int row = g + 8 * hf;
            ps[row * 32 + lm]      = (_Float16)p0[g];
            ps[row * 32 + 16 + lm] = (_Float16)p1[g];
        }
        asm volatile("s_wait_dscnt 0" ::: "memory");
        F16Frag pf;
        #pragma unroll
        for (int v = 0; v < 8; ++v) {
            int kk = hf * 16 + v * 2;
            pf.u[v] = *(const unsigned*)(ps + lm * 32 + kk);
        }

        // ---- O += P(16x32) @ V(32x64): V B-fragments from LDS ----
        {
            F16Frag vf;
            const unsigned* vp = (const unsigned*)(vbuf + (size_t)(0 * 16 + lm) * 32 + hf * 16);
            #pragma unroll
            for (int v = 0; v < 8; ++v) vf.u[v] = vp[v];
            o0 = __builtin_amdgcn_wmma_f32_16x16x32_f16(false, pf.h, false, vf.h,
                                                        (short)0, o0, false, false);
        }
        {
            F16Frag vf;
            const unsigned* vp = (const unsigned*)(vbuf + (size_t)(1 * 16 + lm) * 32 + hf * 16);
            #pragma unroll
            for (int v = 0; v < 8; ++v) vf.u[v] = vp[v];
            o1 = __builtin_amdgcn_wmma_f32_16x16x32_f16(false, pf.h, false, vf.h,
                                                        (short)0, o1, false, false);
        }
        {
            F16Frag vf;
            const unsigned* vp = (const unsigned*)(vbuf + (size_t)(2 * 16 + lm) * 32 + hf * 16);
            #pragma unroll
            for (int v = 0; v < 8; ++v) vf.u[v] = vp[v];
            o2 = __builtin_amdgcn_wmma_f32_16x16x32_f16(false, pf.h, false, vf.h,
                                                        (short)0, o2, false, false);
        }
        {
            F16Frag vf;
            const unsigned* vp = (const unsigned*)(vbuf + (size_t)(3 * 16 + lm) * 32 + hf * 16);
            #pragma unroll
            for (int v = 0; v < 8; ++v) vf.u[v] = vp[v];
            o3 = __builtin_amdgcn_wmma_f32_16x16x32_f16(false, pf.h, false, vf.h,
                                                        (short)0, o3, false, false);
        }
    }

    // ---- epilogue: normalize by l, write f32 out [B,N,64] ----
    float* op = out + ((size_t)b * NTOK + q0r) * DV;
    #pragma unroll
    for (int g = 0; g < 8; ++g) {
        float inv = 1.0f / l[g];
        int row = g + 8 * hf;
        op[row * DV + 0 * 16 + lm] = o0[g] * inv;
        op[row * DV + 1 * 16 + lm] = o1[g] * inv;
        op[row * DV + 2 * 16 + lm] = o2[g] * inv;
        op[row * DV + 3 * 16 + lm] = o3[g] * inv;
    }
}

// ---------------------------------------------------------------------------
extern "C" void kernel_launch(void* const* d_in, const int* in_sizes, int n_in,
                              void* d_out, int out_size, void* d_ws, size_t ws_size,
                              hipStream_t stream)
{
    const float* vectors = (const float*)d_in[0];
    const float* scalars = (const float*)d_in[1];
    const float* Wq      = (const float*)d_in[2];
    const float* Wq_s    = (const float*)d_in[3];
    const float* bq_s    = (const float*)d_in[4];
    const float* Wk      = (const float*)d_in[5];
    const float* Wk_s    = (const float*)d_in[6];
    const float* bk_s    = (const float*)d_in[7];
    const float* Wv      = (const float*)d_in[8];

    _Float16* qb  = (_Float16*)d_ws;                              // [B,N,192]
    _Float16* kbf = qb  + (size_t)BATCH * NTOK * DQK;             // [B,N,192]
    _Float16* vtb = kbf + (size_t)BATCH * NTOK * DQK;             // [B,64,N]
    float* out = (float*)d_out;

    equi_prepass<<<(BATCH * NTOK) / 256, 256, 0, stream>>>(
        vectors, scalars, Wq, Wq_s, bq_s, Wk, Wk_s, bk_s, Wv, qb, kbf, vtb);

    equi_attn<<<BATCH * (NTOK / 128), 256, 0, stream>>>(qb, kbf, vtb, out);
}
